// Light_Spattention_47021301956900
// MI455X (gfx1250) — matrix operand order repaired and checked by reading, not compiled, not run on hardware
//
#include <hip/hip_runtime.h>
#include <cmath>

// ---------------------------------------------------------------------------
// Light_Spattention for MI455X (gfx1250, wave32, WMMA + async-to-LDS).
//
//   out_bt = x_bt @ M_bt  where
//     G_bt   = x_btᵀ x_bt                             (256x256, 2048-deep)
//     S_bth  = K_hᵀ G_bt[:, h-block]                  (64x64)
//     M_bt   = blocks Q_h S_bth * sigmoid(beta_h)/N   (+ sigmoid(alpha_h) diag)
//
// Heavy GEMMs: v_wmma_f32_16x16x32_bf16. Staging: global_load_async_to_lds
// (ASYNCcnt) with double-buffered LDS so DMA overlaps WMMA. Gram contraction
// split into 2 m-slices (partial G buffers) to double block-level parallelism.
// ---------------------------------------------------------------------------

typedef __attribute__((ext_vector_type(16))) __bf16 v16bf;
typedef __attribute__((ext_vector_type(2)))  __bf16 bf16x2;
typedef __attribute__((ext_vector_type(8)))  float  v8f;

#define BT   48          // B*T = 4*12
#define NN   2048        // nodes
#define FD   256         // feature dim
#define HEAD 4
#define HD   64
#define MSLICES 2        // gram k-split
#define MS   (NN / MSLICES)

static __device__ __forceinline__ float sigmoidf_(float v) {
  return 1.0f / (1.0f + __expf(-v));
}

// A-matrix (16x32 bf16) K index for (vgpr v, lane-half):
// lanes 0-15: V0..3 -> K 0..7, V4..7 -> K 16..23 ; lanes 16-31: +8
static __device__ __forceinline__ int a_kidx(int v, int half) {
  return ((v & 3) << 1) + ((v & 4) << 2) + (half << 3);
}
// B-matrix (32x16 bf16) K index: same as A (lanes 0-15: V0..3 -> K 0..7, V4..7 -> K 16..23; lanes 16-31: +8)
static __device__ __forceinline__ int b_kidx(int v, int half) {
  return ((v & 3) << 1) + ((v & 4) << 2) + (half << 3);
}

static __device__ __forceinline__ v8f wmma_bf16(v16bf a, v16bf b, v8f c) {
  return __builtin_amdgcn_wmma_f32_16x16x32_bf16(false, a, false, b, (short)0, c,
                                                 false, false);
}

static __device__ __forceinline__ bf16x2 ld2(const __bf16* p) {
  return *(const bf16x2*)p;   // 4B-aligned b32 load (ds or global)
}

// flat shared-pointer -> LDS byte offset (flat LDS aperture keeps offset in [31:0])
static __device__ __forceinline__ uint32_t lds_off32(const void* p) {
  return (uint32_t)(uintptr_t)p;
}

// gfx1250 async global->LDS copies (tracked by ASYNCcnt, not LOADcnt)
static __device__ __forceinline__ void async_g2l_b32(uint32_t lds_byte, const void* g) {
  asm volatile("global_load_async_to_lds_b32 %0, %1, off"
               :: "v"(lds_byte), "v"(g) : "memory");
}
static __device__ __forceinline__ void async_g2l_b128(uint32_t lds_byte, const void* g) {
  asm volatile("global_load_async_to_lds_b128 %0, %1, off"
               :: "v"(lds_byte), "v"(g) : "memory");
}
static __device__ __forceinline__ void wait_async0() {
  asm volatile("s_wait_asynccnt 0x0" ::: "memory");
}

// ---------------------------------------------------------------------------
// Kernel 1: partial Gram  Gp[ms] = x_sliceᵀ x_slice  (bf16 out, f32 accum)
// grid = BT*4*MSLICES (bt, 64-row strip, m-slice), block = 128 (4 waves).
// Double-buffered async b32 transpose-scatter staging (f32, padded rows).
// ---------------------------------------------------------------------------
__global__ void __launch_bounds__(128)
gram_kernel(const float* __restrict__ x, __bf16* __restrict__ Gp) {
  const int bt     = blockIdx.x >> 3;
  const int strip  = (blockIdx.x >> 1) & 3;
  const int mslice = blockIdx.x & 1;
  const int r0     = strip * 64;
  const int tid    = threadIdx.x;
  const int wave   = tid >> 5;
  const int lane   = tid & 31;
  const int half   = lane >> 4;
  const int j0     = wave * 64;
  const float* xb  = x + (size_t)bt * NN * FD + (size_t)mslice * MS * FD;

  __shared__ float xTf[2][FD][36];        // double-buffered transposed f32
  const uint32_t lbase = lds_off32(&xTf[0][0][0]);
  const uint32_t lbstride = (uint32_t)sizeof(float) * FD * 36;

  auto issue_chunk = [&](int kc, int buf) {
    const int m0 = kc << 5;
    const uint32_t lb = lbase + (uint32_t)buf * lbstride;
    for (int it = tid; it < 8192; it += 128) {
      const int m = it >> 8;               // 0..31
      const int f = it & 255;              // 0..255 (lanes coalesce along f)
      async_g2l_b32(lb + (((uint32_t)(f * 36 + m)) << 2),
                    xb + (size_t)(m0 + m) * FD + f);
    }
  };

  v8f acc[4][4] = {};
  const int NCH = MS / 32;                 // 32 chunks

  issue_chunk(0, 0);
  wait_async0();
  __syncthreads();

  for (int kc = 0; kc < NCH; ++kc) {
    const int cur = kc & 1;
    if (kc + 1 < NCH) issue_chunk(kc + 1, cur ^ 1);   // DMA overlaps compute

    v16bf afr[4];
#pragma unroll
    for (int ti = 0; ti < 4; ++ti) {
      const int f = r0 + ti * 16 + (lane & 15);
#pragma unroll
      for (int v = 0; v < 8; ++v) {
        float2 p = *(const float2*)&xTf[cur][f][a_kidx(v, half)];
        afr[ti][2 * v] = (__bf16)p.x; afr[ti][2 * v + 1] = (__bf16)p.y;
      }
    }
#pragma unroll
    for (int tj = 0; tj < 4; ++tj) {
      const int j = j0 + tj * 16 + (lane & 15);
      v16bf bfr;
#pragma unroll
      for (int v = 0; v < 8; ++v) {
        float2 p = *(const float2*)&xTf[cur][j][b_kidx(v, half)];
        bfr[2 * v] = (__bf16)p.x; bfr[2 * v + 1] = (__bf16)p.y;
      }
#pragma unroll
      for (int ti = 0; ti < 4; ++ti)
        acc[ti][tj] = wmma_bf16(afr[ti], bfr, acc[ti][tj]);
    }

    wait_async0();       // next chunk's DMA done (this wave)
    __syncthreads();     // all waves done reading cur; next buffer visible
  }

  __bf16* Gb = Gp + (size_t)mslice * BT * FD * FD + ((size_t)bt << 16);
#pragma unroll
  for (int ti = 0; ti < 4; ++ti)
#pragma unroll
    for (int tj = 0; tj < 4; ++tj)
#pragma unroll
      for (int v = 0; v < 8; ++v) {
        const int fi = r0 + ti * 16 + v + (half << 3);
        const int fj = j0 + tj * 16 + (lane & 15);
        Gb[(size_t)fi * FD + fj] = (__bf16)acc[ti][tj][v];
      }
}

// ---------------------------------------------------------------------------
// Kernel 2: per (bt, h):  S = K_hᵀ (G0+G1)_h (64x64),  M_h = Q_h S * cb
// (+ sa diag), stored transposed MT[j][f].  grid = BT*HEAD, block = 128.
// ---------------------------------------------------------------------------
__global__ void __launch_bounds__(128)
build_m_kernel(const float* __restrict__ Qg, const float* __restrict__ Kg,
               const __bf16* __restrict__ Gp, const float* __restrict__ alpha,
               const float* __restrict__ beta, __bf16* __restrict__ MT) {
  const int bt   = blockIdx.x >> 2;
  const int h    = blockIdx.x & 3;
  const int tid  = threadIdx.x;
  const int wave = tid >> 5;
  const int lane = tid & 31;
  const int half = lane >> 4;
  const __bf16* G0b = Gp + ((size_t)bt << 16);
  const __bf16* G1b = G0b + (size_t)BT * FD * FD;

  __shared__ __bf16 KT[64][132];   // KT[s][f_local] = K[f, 64h+s]
  __shared__ __bf16 GT[64][132];   // GT[d][f_local] = (G0+G1)[f, 64h+d]
  __shared__ __bf16 ST[64][68];    // ST[d][s]       = S[s, d]

  // ---- phase 1: S = K_hᵀ G_h ; wave w owns S rows [16w,16w+16) ----
  v8f sacc[4] = {};
  for (int fc = 0; fc < 2; ++fc) {
    const int fbase = fc << 7;
    __syncthreads();
    for (int it = tid; it < 2048; it += 128) {
      const int f = it >> 4, sq = (it & 15) << 2;
      float4 vv = *(const float4*)(Kg + (size_t)(fbase + f) * FD + h * HD + sq);
      KT[sq + 0][f] = (__bf16)vv.x; KT[sq + 1][f] = (__bf16)vv.y;
      KT[sq + 2][f] = (__bf16)vv.z; KT[sq + 3][f] = (__bf16)vv.w;
    }
    for (int it = tid; it < 2048; it += 128) {
      const int f = it >> 4, dq = (it & 15) << 2;
      const size_t off = (size_t)(fbase + f) * FD + h * HD + dq;
      const __bf16* g0 = G0b + off;
      const __bf16* g1 = G1b + off;
#pragma unroll
      for (int e = 0; e < 4; ++e)
        GT[dq + e][f] = (__bf16)((float)g0[e] + (float)g1[e]);
    }
    __syncthreads();
#pragma unroll
    for (int kc = 0; kc < 4; ++kc) {
      const int k0 = kc << 5;
      v16bf afr;
      const int srow = (wave << 4) + (lane & 15);
#pragma unroll
      for (int v = 0; v < 8; ++v) {
        bf16x2 p = ld2(&KT[srow][k0 + a_kidx(v, half)]);
        afr[2 * v] = p[0]; afr[2 * v + 1] = p[1];
      }
#pragma unroll
      for (int tj = 0; tj < 4; ++tj) {
        const int d = tj * 16 + (lane & 15);
        v16bf bfr;
#pragma unroll
        for (int v = 0; v < 8; ++v) {
          bf16x2 p = ld2(&GT[d][k0 + b_kidx(v, half)]);
          bfr[2 * v] = p[0]; bfr[2 * v + 1] = p[1];
        }
        sacc[tj] = wmma_bf16(afr, bfr, sacc[tj]);
      }
    }
  }
  // write S transposed to LDS for phase 2's B fragments
#pragma unroll
  for (int tj = 0; tj < 4; ++tj)
#pragma unroll
    for (int v = 0; v < 8; ++v) {
      const int s = (wave << 4) + v + (half << 3);
      const int d = tj * 16 + (lane & 15);
      ST[d][s] = (__bf16)sacc[tj][v];
    }
  __syncthreads();

  // ---- phase 2: M_h = Q_h S ; wave w owns rows f in [64w, 64w+64) ----
  const float sa = sigmoidf_(alpha[h]);
  const float cb = sigmoidf_(beta[h]) * (1.0f / (float)NN);
  v8f macc[4][4] = {};
#pragma unroll
  for (int kc = 0; kc < 2; ++kc) {
    const int k0 = kc << 5;
    v16bf afr[4];
#pragma unroll
    for (int ti = 0; ti < 4; ++ti) {
      const int f = (wave << 6) + ti * 16 + (lane & 15);
      const float* qp = Qg + (size_t)f * FD + h * HD;
#pragma unroll
      for (int v = 0; v < 8; ++v) {
        float2 qq = *(const float2*)(qp + k0 + a_kidx(v, half));
        afr[ti][2 * v] = (__bf16)qq.x; afr[ti][2 * v + 1] = (__bf16)qq.y;
      }
    }
#pragma unroll
    for (int tj = 0; tj < 4; ++tj) {
      const int d = tj * 16 + (lane & 15);
      v16bf bfr;
#pragma unroll
      for (int v = 0; v < 8; ++v) {
        bf16x2 p = ld2(&ST[d][k0 + b_kidx(v, half)]);
        bfr[2 * v] = p[0]; bfr[2 * v + 1] = p[1];
      }
#pragma unroll
      for (int ti = 0; ti < 4; ++ti)
        macc[ti][tj] = wmma_bf16(afr[ti], bfr, macc[ti][tj]);
    }
  }

  __bf16* MTb = MT + ((size_t)bt << 16);
#pragma unroll
  for (int ti = 0; ti < 4; ++ti)
#pragma unroll
    for (int tj = 0; tj < 4; ++tj)
#pragma unroll
      for (int v = 0; v < 8; ++v) {
        const int f = (wave << 6) + ti * 16 + v + (half << 3);
        const int j = h * HD + tj * 16 + (lane & 15);
        float val = macc[ti][tj][v] * cb + ((f == j) ? sa : 0.0f);
        MTb[(size_t)j * FD + f] = (__bf16)val;   // stored transposed
      }
}

// ---------------------------------------------------------------------------
// Kernel 3: out_bt = x_bt @ M_bt.  grid = BT*32 (64-row strips), block = 128.
// Double-buffered async b128 staging of the f32 x chunk (row-major);
// B fragments are contiguous bf16-pair global loads from transposed M.
// ---------------------------------------------------------------------------
__global__ void __launch_bounds__(128)
out_kernel(const float* __restrict__ x, const __bf16* __restrict__ MT,
           float* __restrict__ out) {
  const int bt   = blockIdx.x >> 5;
  const int n0   = (blockIdx.x & 31) * 64;
  const int tid  = threadIdx.x;
  const int wave = tid >> 5;
  const int lane = tid & 31;
  const int half = lane >> 4;
  const float*  xb  = x   + (size_t)bt * NN * FD;
  const __bf16* MTb = MT  + ((size_t)bt << 16);
  float*        ob  = out + (size_t)bt * NN * FD;

  __shared__ float xsf[2][64][36];        // double-buffered row-major f32 chunk
  const uint32_t lbase = lds_off32(&xsf[0][0][0]);
  const uint32_t lbstride = (uint32_t)sizeof(float) * 64 * 36;

  auto issue_chunk = [&](int kc, int buf) {
    const int k0 = kc << 5;
    const uint32_t lb = lbase + (uint32_t)buf * lbstride;
    for (int it = tid; it < 512; it += 128) {
      const int m = it >> 3, q = (it & 7) << 2;
      async_g2l_b128(lb + (((uint32_t)(m * 36 + q)) << 2),
                     xb + (size_t)(n0 + m) * FD + k0 + q);
    }
  };

  v8f acc[4][4] = {};
  const int NCH = FD / 32;                 // 8 chunks

  issue_chunk(0, 0);
  wait_async0();
  __syncthreads();

  for (int kc = 0; kc < NCH; ++kc) {
    const int cur = kc & 1;
    const int k0  = kc << 5;
    if (kc + 1 < NCH) issue_chunk(kc + 1, cur ^ 1);

    v16bf afr[4];
#pragma unroll
    for (int ti = 0; ti < 4; ++ti) {
      const int m = ti * 16 + (lane & 15);
#pragma unroll
      for (int v = 0; v < 8; ++v) {
        float2 p = *(const float2*)&xsf[cur][m][a_kidx(v, half)];
        afr[ti][2 * v] = (__bf16)p.x; afr[ti][2 * v + 1] = (__bf16)p.y;
      }
    }
#pragma unroll
    for (int tj = 0; tj < 4; ++tj) {
      const int j = (wave << 6) + tj * 16 + (lane & 15);
      const __bf16* mp = MTb + (size_t)j * FD + k0;
      v16bf bfr;
#pragma unroll
      for (int v = 0; v < 8; ++v) {
        bf16x2 p = ld2(mp + b_kidx(v, half));
        bfr[2 * v] = p[0]; bfr[2 * v + 1] = p[1];
      }
#pragma unroll
      for (int ti = 0; ti < 4; ++ti)
        acc[ti][tj] = wmma_bf16(afr[ti], bfr, acc[ti][tj]);
    }

    wait_async0();
    __syncthreads();
  }

#pragma unroll
  for (int ti = 0; ti < 4; ++ti)
#pragma unroll
    for (int tj = 0; tj < 4; ++tj)
#pragma unroll
      for (int v = 0; v < 8; ++v) {
        const int n = n0 + ti * 16 + v + (half << 3);
        const int j = (wave << 6) + tj * 16 + (lane & 15);
        ob[(size_t)n * FD + j] = acc[ti][tj][v];
      }
}

// ---------------------------------------------------------------------------
extern "C" void kernel_launch(void* const* d_in, const int* in_sizes, int n_in,
                              void* d_out, int out_size, void* d_ws, size_t ws_size,
                              hipStream_t stream) {
  (void)in_sizes; (void)n_in; (void)out_size; (void)ws_size;
  const float* x     = (const float*)d_in[0];
  const float* Q     = (const float*)d_in[1];
  const float* K     = (const float*)d_in[2];
  const float* alpha = (const float*)d_in[3];
  const float* beta  = (const float*)d_in[4];
  float* out = (float*)d_out;

  // workspace: Gp[2] (bf16, 2*48*256*256) | MT (bf16, 48*256*256)  => 18.9 MB
  const size_t gbytes = (size_t)BT * FD * FD * sizeof(unsigned short);
  __bf16* Gp = (__bf16*)d_ws;
  __bf16* MT = (__bf16*)((char*)d_ws + MSLICES * gbytes);

  gram_kernel   <<<BT * 4 * MSLICES, 128, 0, stream>>>(x, Gp);
  build_m_kernel<<<BT * HEAD,        128, 0, stream>>>(Q, K, Gp, alpha, beta, MT);
  out_kernel    <<<BT * 32,          128, 0, stream>>>(x, MT, out);
}
